// DoubleStreamBlock_84756884620116
// MI455X (gfx1250) — matrix operand order repaired and checked by reading, not compiled, not run on hardware
//
#include <hip/hip_runtime.h>
#include <cstddef>
#include <cstdint>

// ---------------------------------------------------------------------------
// DoubleStreamBlock (Flux MMDiT) for MI455X / gfx1250, wave32 + WMMA bf16.
// B=1, LIMG=2048, LTXT=256, H=3072, NH=24, HD=128, MLP=12288, LTOT=2304.
// ---------------------------------------------------------------------------

#define DEVFN static __device__ __forceinline__

typedef unsigned short u16;
typedef __bf16 bf16;
typedef bf16  v16bf  __attribute__((ext_vector_type(16)));
typedef float v8f    __attribute__((ext_vector_type(8)));
typedef u16   u16x4  __attribute__((ext_vector_type(4)));
typedef u16   u16x8  __attribute__((ext_vector_type(8)));
typedef u16   u16x16 __attribute__((ext_vector_type(16)));
typedef float fx4    __attribute__((ext_vector_type(4)));

constexpr int Hdim  = 3072;
constexpr int NH_   = 24;
constexpr int HD_   = 128;
constexpr int LIMG_ = 2048;
constexpr int LTXT_ = 256;
constexpr int LTOT_ = 2304;
constexpr int MLP_  = 12288;
constexpr int SIXH  = 6 * Hdim;

// float -> bf16 bits, round-to-nearest-even
DEVFN u16 f2bf(float f) {
    unsigned u = __builtin_bit_cast(unsigned, f);
    unsigned r = u + 0x7FFFu + ((u >> 16) & 1u);
    return (u16)(r >> 16);
}

// build a v16bf fragment from two 16-byte LDS/global chunks
DEVFN v16bf ld_frag2(const u16* p0, const u16* p1) {
    u16x8 lo = *(const u16x8*)p0;
    u16x8 hi = *(const u16x8*)p1;
    u16x16 c = __builtin_shufflevector(lo, hi, 0,1,2,3,4,5,6,7,8,9,10,11,12,13,14,15);
    return __builtin_bit_cast(v16bf, c);
}

DEVFN v8f wmma_bf16(v16bf a, v16bf b, v8f c) {
    return __builtin_amdgcn_wmma_f32_16x16x32_bf16(
        /*neg_a=*/false, a, /*neg_b=*/false, b,
        /*c_mod=*/(short)0, c, /*reuse_a=*/false, /*reuse_b=*/false);
}

DEVFN float gelu_tanh(float x) {
    float x3 = x * x * x;
    return 0.5f * x * (1.f + tanhf(0.7978845608028654f * (x + 0.044715f * x3)));
}

// generic pointer to __shared__ -> LDS byte address (low 32 bits of flat addr)
DEVFN unsigned lds_addr32(const void* p) {
    return (unsigned)(uintptr_t)p;
}

// async global->LDS 16-byte copy (GVS mode), tracked by ASYNCcnt
DEVFN void async_g2l_b128(unsigned lds_off, const void* sbase, unsigned voff) {
    asm volatile("global_load_async_to_lds_b128 %0, %1, %2"
                 :: "v"(lds_off), "v"(voff), "s"(sbase)
                 : "memory");
}

DEVFN void wait_asynccnt0() {
    asm volatile("s_wait_asynccnt 0x0" ::: "memory");
}

// block-wide sum over (nwaves*32) threads, sbuf >= 8 floats
DEVFN float block_sum(float v, float* sbuf, int nwaves) {
    int lane = threadIdx.x & 31, wid = threadIdx.x >> 5;
#pragma unroll
    for (int off = 16; off; off >>= 1) v += __shfl_xor(v, off, 32);
    if (lane == 0) sbuf[wid] = v;
    __syncthreads();
    float t = (threadIdx.x < (unsigned)nwaves) ? sbuf[threadIdx.x] : 0.f;
    if (wid == 0) {
#pragma unroll
        for (int off = 4; off; off >>= 1) t += __shfl_xor(t, off, 32);
        if (lane == 0) sbuf[0] = t;
    }
    __syncthreads();
    float r = sbuf[0];
    __syncthreads();
    return r;
}

// ---------------------------------------------------------------------------
// 1) modulation GEMV: out[col] = silu(vec) . W[:,col] + b[col]   (N = 18432)
// ---------------------------------------------------------------------------
__global__ __launch_bounds__(256) void mod_gemv(
    const float* __restrict__ vec, const float* __restrict__ W,
    const float* __restrict__ b, float* __restrict__ out)
{
    __shared__ float sv[Hdim];
    int tid = threadIdx.x;
    for (int i = tid; i < Hdim; i += 256) {
        float v = vec[i];
        sv[i] = v / (1.f + __expf(-v));
    }
    __syncthreads();
    int col = blockIdx.x * 256 + tid;
    float acc = 0.f;
    for (int k = 0; k < Hdim; ++k)
        acc = fmaf(sv[k], W[(size_t)k * SIXH + col], acc);
    out[col] = acc + b[col];
}

// ---------------------------------------------------------------------------
// 2) fused LayerNorm + modulation: out = (1+sc)*((x-m)*rinv*s + b) + sh
// ---------------------------------------------------------------------------
__global__ __launch_bounds__(256) void ln_mod(
    const float* __restrict__ x, const float* __restrict__ s,
    const float* __restrict__ bb, const float* __restrict__ sc,
    const float* __restrict__ sh, float* __restrict__ out)
{
    __shared__ float sbuf[8];
    int row = blockIdx.x, tid = threadIdx.x;
    const float* xr = x + (size_t)row * Hdim;
    float s1 = 0.f, s2 = 0.f;
    for (int c = tid; c < Hdim; c += 256) {
        float v = xr[c];
        s1 += v; s2 += v * v;
    }
    s1 = block_sum(s1, sbuf, 8);
    s2 = block_sum(s2, sbuf, 8);
    float mean = s1 * (1.f / Hdim);
    float var  = s2 * (1.f / Hdim) - mean * mean;
    float rinv = rsqrtf(var + 1e-6f);
    float* orow = out + (size_t)row * Hdim;
    for (int c = tid; c < Hdim; c += 256) {
        float nx = (xr[c] - mean) * rinv * s[c] + bb[c];
        orow[c] = (1.f + sc[c]) * nx + sh[c];
    }
}

// ---------------------------------------------------------------------------
// 3) WMMA bf16 GEMM: out[M,N] = epi(A[M,K] * W[K,N] + bias)
//    EPI 0: +bias   EPI 1: gelu(+bias)   EPI 2: res + gate*(+bias)
//    Block tile 128x128, K-step 32, 8 waves -> 64x32 per wave (4x2 accums).
//    Register-pipelined double-buffered LDS: one barrier per K-step, global
//    loads of tile k+1 overlap WMMAs on tile k.
// ---------------------------------------------------------------------------
template <int EPI>
__global__ __launch_bounds__(256) void gemm_bf16(
    const float* __restrict__ A, const float* __restrict__ W,
    const float* __restrict__ bias, const float* __restrict__ gate,
    const float* __restrict__ res, float* __restrict__ out,
    int M, int N, int K)
{
    (void)M;
    __shared__ u16 As[2][128 * 40];   // [row][k], stride 40 halves
    __shared__ u16 Bs[2][128 * 40];   // [n][k] transposed, stride 40 halves

    const int tid  = threadIdx.x;
    const int lane = tid & 31;
    const int ln   = lane & 15, hs = lane >> 4;
    const int wid  = tid >> 5;
    const int wm   = wid >> 2;   // 0..1 : 64-row slice
    const int wn   = wid & 3;    // 0..3 : 32-col slice
    const int bm   = blockIdx.y * 128;
    const int bn   = blockIdx.x * 128;

    v8f acc[4][2];
#pragma unroll
    for (int i = 0; i < 4; ++i)
#pragma unroll
        for (int j = 0; j < 2; ++j) {
            v8f z = {0.f, 0.f, 0.f, 0.f, 0.f, 0.f, 0.f, 0.f};
            acc[i][j] = z;
        }

    const int nk = K >> 5;
    fx4 ra[4], rw[4];

    // preload tile 0 into registers
#pragma unroll
    for (int p = 0; p < 4; ++p) {
        int idx = p * 256 + tid;
        ra[p] = *(const fx4*)&A[(size_t)(bm + (idx >> 3)) * K + ((idx & 7) << 2)];
        rw[p] = *(const fx4*)&W[(size_t)(idx >> 5) * N + bn + ((idx & 31) << 2)];
    }

    int buf = 0;
    for (int kt = 0; kt < nk; ++kt) {
        // convert + store current tile registers into LDS[buf]
#pragma unroll
        for (int p = 0; p < 4; ++p) {
            int idx = p * 256 + tid;
            int row = idx >> 3, c4 = (idx & 7) << 2;
            u16x4 h;
            h[0] = f2bf(ra[p][0]); h[1] = f2bf(ra[p][1]);
            h[2] = f2bf(ra[p][2]); h[3] = f2bf(ra[p][3]);
            *(u16x4*)&As[buf][row * 40 + c4] = h;
            int kk = idx >> 5, n4 = (idx & 31) << 2;
            Bs[buf][(n4 + 0) * 40 + kk] = f2bf(rw[p][0]);
            Bs[buf][(n4 + 1) * 40 + kk] = f2bf(rw[p][1]);
            Bs[buf][(n4 + 2) * 40 + kk] = f2bf(rw[p][2]);
            Bs[buf][(n4 + 3) * 40 + kk] = f2bf(rw[p][3]);
        }
        __syncthreads();

        // issue global loads for tile kt+1 (overlap with WMMA below)
        if (kt + 1 < nk) {
#pragma unroll
            for (int p = 0; p < 4; ++p) {
                int idx = p * 256 + tid;
                ra[p] = *(const fx4*)&A[(size_t)(bm + (idx >> 3)) * K + (kt + 1) * 32 + ((idx & 7) << 2)];
                rw[p] = *(const fx4*)&W[(size_t)((kt + 1) * 32 + (idx >> 5)) * N + bn + ((idx & 31) << 2)];
            }
            if (kt + 2 < nk)  // L2 hint two tiles ahead (global_prefetch_b8)
                __builtin_prefetch(&W[(size_t)((kt + 2) * 32 + (tid >> 3)) * N + bn + ((tid & 7) << 4)], 0, 1);
        }

        v16bf bfrag[2], afrag[4];
#pragma unroll
        for (int fn = 0; fn < 2; ++fn) {
            const u16* p0 = &Bs[buf][(wn * 32 + fn * 16 + ln) * 40 + 16 * hs];
            bfrag[fn] = ld_frag2(p0, p0 + 8);     // B: K = 16*hs + h
        }
#pragma unroll
        for (int fm = 0; fm < 4; ++fm) {
            const u16* p0 = &As[buf][(wm * 64 + fm * 16 + ln) * 40 + 16 * hs];
            afrag[fm] = ld_frag2(p0, p0 + 8);     // A: K = 16*hs + h
        }
#pragma unroll
        for (int fm = 0; fm < 4; ++fm)
#pragma unroll
            for (int fn = 0; fn < 2; ++fn)
                acc[fm][fn] = wmma_bf16(afrag[fm], bfrag[fn], acc[fm][fn]);

        buf ^= 1;
    }

    // epilogue
    const int crow0 = bm + wm * 64;
#pragma unroll
    for (int fm = 0; fm < 4; ++fm) {
#pragma unroll
        for (int fn = 0; fn < 2; ++fn) {
            int gc = bn + wn * 32 + fn * 16 + ln;
            float bv = bias[gc];
#pragma unroll
            for (int r = 0; r < 8; ++r) {
                int gr = crow0 + fm * 16 + 8 * hs + r;
                size_t o = (size_t)gr * N + gc;
                float v = acc[fm][fn][r] + bv;
                if constexpr (EPI == 1) v = gelu_tanh(v);
                if constexpr (EPI == 2) v = res[o] + gate[gc] * v;
                out[o] = v;
            }
        }
    }
}

// ---------------------------------------------------------------------------
// 4) QKV head prep: RMS-norm q/k, RoPE, fold 1/sqrt(HD), store bf16
//    qkv row layout [l][3*H]; outputs [head][abs_pos][128] bf16.
// ---------------------------------------------------------------------------
__global__ __launch_bounds__(128) void qkv_prep(
    const float* __restrict__ qkv, const float* __restrict__ qn,
    const float* __restrict__ kn, const float* __restrict__ pe,
    u16* __restrict__ qh, u16* __restrict__ kh, u16* __restrict__ vh,
    int pos_off)
{
    __shared__ float sbuf[8];
    int blk = blockIdx.x;
    int l = blk / NH_, head = blk % NH_;
    int d = threadIdx.x;
    size_t base = (size_t)l * (3 * Hdim) + head * HD_ + d;
    float q = qkv[base];
    float k = qkv[base + Hdim];
    float v = qkv[base + 2 * Hdim];

    float qs = block_sum(q * q, sbuf, 4);
    float ks = block_sum(k * k, sbuf, 4);
    q *= rsqrtf(qs * (1.f / HD_) + 1e-6f) * qn[d];
    k *= rsqrtf(ks * (1.f / HD_) + 1e-6f) * kn[d];

    int pos = pos_off + l;
    // RoPE: pairs (2j, 2j+1); pe shape [pos][64][2][2]
    const float* per = pe + (size_t)pos * 256 + (d >> 1) * 4 + (d & 1) * 2;
    float qp = __shfl_xor(q, 1, 32);
    float kp = __shfl_xor(k, 1, 32);
    float q0 = (d & 1) ? qp : q, q1 = (d & 1) ? q : qp;
    float k0 = (d & 1) ? kp : k, k1 = (d & 1) ? k : kp;
    float qo = per[0] * q0 + per[1] * q1;
    float ko = per[0] * k0 + per[1] * k1;
    qo *= 0.08838834764831845f;   // 1/sqrt(128) folded into q

    size_t hidx = ((size_t)head * LTOT_ + pos) * HD_ + d;
    qh[hidx] = f2bf(qo);
    kh[hidx] = f2bf(ko);
    vh[hidx] = f2bf(v);
}

// ---------------------------------------------------------------------------
// 5) Flash attention: grid (LTOT/128, NH), 8 waves x 16 q-rows,
//    72 chunks of 32 keys, online softmax, WMMA bf16 for QK^T and PV.
//    K chunk staged via GLOBAL_LOAD_ASYNC_TO_LDS_B128 (ASYNCcnt path).
// ---------------------------------------------------------------------------
__global__ __launch_bounds__(256) void attn_kernel(
    const u16* __restrict__ qh, const u16* __restrict__ kh,
    const u16* __restrict__ vh, float* __restrict__ out)
{
    __shared__ u16 Ks[32 * 136];     // [key][d]
    __shared__ u16 Vs[128 * 40];     // [d][key] (transposed)
    __shared__ u16 Ps[8 * 16 * 40];  // per-wave 16x32 P tile

    const int head = blockIdx.y, qtile = blockIdx.x;
    const int tid = threadIdx.x, lane = tid & 31, wid = tid >> 5;
    const int ln = lane & 15, hs = lane >> 4;

    const u16* qp = qh + (size_t)head * LTOT_ * HD_;
    const u16* kp = kh + (size_t)head * LTOT_ * HD_;
    const u16* vp = vh + (size_t)head * LTOT_ * HD_;

    const int qbase = qtile * 128 + wid * 16;
    v16bf qa[4];
#pragma unroll
    for (int kk = 0; kk < 4; ++kk) {
        const u16* p = qp + (size_t)(qbase + ln) * HD_ + kk * 32 + 16 * hs;
        qa[kk] = ld_frag2(p, p + 8);
    }

    v8f o[8];
#pragma unroll
    for (int i = 0; i < 8; ++i) {
        v8f z = {0.f, 0.f, 0.f, 0.f, 0.f, 0.f, 0.f, 0.f};
        o[i] = z;
    }
    float m[8], l[8];
#pragma unroll
    for (int r = 0; r < 8; ++r) { m[r] = -1e30f; l[r] = 0.f; }

    for (int c = 0; c < LTOT_ / 32; ++c) {
        __syncthreads();
        // stage K chunk [32][128] via async global->LDS (bf16, layout match)
#pragma unroll
        for (int p = 0; p < 2; ++p) {
            int idx = p * 256 + tid;
            int key = idx >> 4, c8 = (idx & 15) << 3;
            unsigned loff = lds_addr32(&Ks[key * 136 + c8]);
            unsigned goff = (unsigned)(((c * 32 + key) * HD_ + c8) * 2);
            async_g2l_b128(loff, kp, goff);
        }
        // stage V chunk transposed [128][32] (manual: needs transpose)
#pragma unroll
        for (int p = 0; p < 4; ++p) {
            int idx = p * 256 + tid;
            int key = idx >> 5, d4 = (idx & 31) << 2;
            u16x4 f = *(const u16x4*)&vp[(size_t)(c * 32 + key) * HD_ + d4];
            Vs[(d4 + 0) * 40 + key] = f[0];
            Vs[(d4 + 1) * 40 + key] = f[1];
            Vs[(d4 + 2) * 40 + key] = f[2];
            Vs[(d4 + 3) * 40 + key] = f[3];
        }
        wait_asynccnt0();
        __syncthreads();

        // S = Q * K^T for this wave's 16 rows, 32 keys
        v8f s0 = {0.f,0.f,0.f,0.f,0.f,0.f,0.f,0.f};
        v8f s1 = {0.f,0.f,0.f,0.f,0.f,0.f,0.f,0.f};
#pragma unroll
        for (int kk = 0; kk < 4; ++kk) {
            const u16* b0 = &Ks[(ln) * 136 + kk * 32 + 16 * hs];
            const u16* b1 = &Ks[(16 + ln) * 136 + kk * 32 + 16 * hs];
            s0 = wmma_bf16(qa[kk], ld_frag2(b0, b0 + 8), s0);
            s1 = wmma_bf16(qa[kk], ld_frag2(b1, b1 + 8), s1);
        }

        // online softmax (row = r + 8*hs, cols across 16 lanes)
        float corr[8];
        u16* Pw = &Ps[wid * 16 * 40];
#pragma unroll
        for (int r = 0; r < 8; ++r) {
            float v0 = s0[r], v1 = s1[r];
            float mx = fmaxf(v0, v1);
#pragma unroll
            for (int off = 8; off; off >>= 1) mx = fmaxf(mx, __shfl_xor(mx, off, 32));
            float mn = fmaxf(m[r], mx);
            float p0 = __expf(v0 - mn), p1 = __expf(v1 - mn);
            float ps = p0 + p1;
#pragma unroll
            for (int off = 8; off; off >>= 1) ps += __shfl_xor(ps, off, 32);
            float cr = __expf(m[r] - mn);
            l[r] = l[r] * cr + ps;
            m[r] = mn;
            corr[r] = cr;
            Pw[(r + 8 * hs) * 40 + ln]      = f2bf(p0);
            Pw[(r + 8 * hs) * 40 + 16 + ln] = f2bf(p1);
        }
#pragma unroll
        for (int fc = 0; fc < 8; ++fc) {
            v8f t = o[fc];
#pragma unroll
            for (int r = 0; r < 8; ++r) t[r] *= corr[r];
            o[fc] = t;
        }
        // P as A-fragment (16x32), V as B-fragments -> O += P*V
        const u16* pp = &Pw[ln * 40 + 16 * hs];
        v16bf pa = ld_frag2(pp, pp + 8);
#pragma unroll
        for (int fc = 0; fc < 8; ++fc) {
            const u16* vv = &Vs[(fc * 16 + ln) * 40 + 16 * hs];
            o[fc] = wmma_bf16(pa, ld_frag2(vv, vv + 8), o[fc]);
        }
    }

    // write normalized output: out[row][head*128 + col]
#pragma unroll
    for (int fc = 0; fc < 8; ++fc) {
#pragma unroll
        for (int r = 0; r < 8; ++r) {
            int row = qtile * 128 + wid * 16 + 8 * hs + r;
            int col = head * HD_ + fc * 16 + ln;
            out[(size_t)row * Hdim + col] = o[fc][r] / l[r];
        }
    }
}

// ---------------------------------------------------------------------------
// host-side orchestration
// ---------------------------------------------------------------------------
extern "C" void kernel_launch(void* const* d_in, const int* in_sizes, int n_in,
                              void* d_out, int out_size, void* d_ws, size_t ws_size,
                              hipStream_t stream)
{
    (void)in_sizes; (void)n_in; (void)out_size; (void)ws_size;

    const float* img = (const float*)d_in[0];
    const float* txt = (const float*)d_in[1];
    const float* vec = (const float*)d_in[2];
    const float* pe  = (const float*)d_in[3];
    // per-stream params: 0 mod_w 1 mod_b 2 ln1_s 3 ln1_b 4 qkv_w 5 qkv_b 6 qn
    // 7 kn 8 proj_w 9 proj_b 10 ln2_s 11 ln2_b 12 mlp_w1 13 mlp_b1 14 mlp_w2 15 mlp_b2
    auto Pi = [&](int i) { return (const float*)d_in[4 + i]; };
    auto Pt = [&](int i) { return (const float*)d_in[20 + i]; };

    char* ws = (char*)d_ws;
    size_t off = 0;
    auto alloc = [&](size_t bytes) -> char* {
        char* p = ws + off;
        off = (off + bytes + 255) & ~(size_t)255;
        return p;
    };

    float* mod_i = (float*)alloc((size_t)SIXH * 4);
    float* mod_t = (float*)alloc((size_t)SIXH * 4);
    float* xm_i  = (float*)alloc((size_t)LIMG_ * Hdim * 4);
    float* xm_t  = (float*)alloc((size_t)LTXT_ * Hdim * 4);
    // region R1 reused: QKV output (x9216) first, then GELU activations (x12288)
    float* R1    = (float*)alloc((size_t)(LIMG_ + LTXT_) * MLP_ * 4);
    float* qkv_i = R1;
    float* qkv_t = R1 + (size_t)LIMG_ * (3 * Hdim);
    float* act_i = R1;
    float* act_t = R1 + (size_t)LIMG_ * MLP_;
    u16* qh = (u16*)alloc((size_t)NH_ * LTOT_ * HD_ * 2);
    u16* kh = (u16*)alloc((size_t)NH_ * LTOT_ * HD_ * 2);
    u16* vh = (u16*)alloc((size_t)NH_ * LTOT_ * HD_ * 2);
    float* attn = (float*)alloc((size_t)LTOT_ * Hdim * 4);
    float* x1_i = (float*)alloc((size_t)LIMG_ * Hdim * 4);
    float* x1_t = (float*)alloc((size_t)LTXT_ * Hdim * 4);

    float* out_img = (float*)d_out;
    float* out_txt = out_img + (size_t)LIMG_ * Hdim;

    // 1) modulation vectors
    mod_gemv<<<SIXH / 256, 256, 0, stream>>>(vec, Pi(0), Pi(1), mod_i);
    mod_gemv<<<SIXH / 256, 256, 0, stream>>>(vec, Pt(0), Pt(1), mod_t);

    // 2) LN1 + (1+sc1)*x + sh1
    ln_mod<<<LIMG_, 256, 0, stream>>>(img, Pi(2), Pi(3), mod_i + Hdim, mod_i, xm_i);
    ln_mod<<<LTXT_, 256, 0, stream>>>(txt, Pt(2), Pt(3), mod_t + Hdim, mod_t, xm_t);

    // 3) QKV GEMMs (bias epilogue)
    gemm_bf16<0><<<dim3(9216 / 128, LIMG_ / 128), 256, 0, stream>>>(
        xm_i, Pi(4), Pi(5), nullptr, nullptr, qkv_i, LIMG_, 9216, Hdim);
    gemm_bf16<0><<<dim3(9216 / 128, LTXT_ / 128), 256, 0, stream>>>(
        xm_t, Pt(4), Pt(5), nullptr, nullptr, qkv_t, LTXT_, 9216, Hdim);

    // 4) head split + RMS + RoPE -> bf16 q/k/v (txt at pos 0, img at pos 256)
    qkv_prep<<<LTXT_ * NH_, 128, 0, stream>>>(qkv_t, Pt(6), Pt(7), pe, qh, kh, vh, 0);
    qkv_prep<<<LIMG_ * NH_, 128, 0, stream>>>(qkv_i, Pi(6), Pi(7), pe, qh, kh, vh, LTXT_);

    // 5) joint flash attention
    attn_kernel<<<dim3(LTOT_ / 128, NH_), 256, 0, stream>>>(qh, kh, vh, attn);

    // 6) proj + gated residual
    gemm_bf16<2><<<dim3(Hdim / 128, LIMG_ / 128), 256, 0, stream>>>(
        attn + (size_t)LTXT_ * Hdim, Pi(8), Pi(9), mod_i + 2 * Hdim, img, x1_i,
        LIMG_, Hdim, Hdim);
    gemm_bf16<2><<<dim3(Hdim / 128, LTXT_ / 128), 256, 0, stream>>>(
        attn, Pt(8), Pt(9), mod_t + 2 * Hdim, txt, x1_t, LTXT_, Hdim, Hdim);

    // 7) LN2 + (1+sc2)*x + sh2
    ln_mod<<<LIMG_, 256, 0, stream>>>(x1_i, Pi(10), Pi(11), mod_i + 4 * Hdim, mod_i + 3 * Hdim, xm_i);
    ln_mod<<<LTXT_, 256, 0, stream>>>(x1_t, Pt(10), Pt(11), mod_t + 4 * Hdim, mod_t + 3 * Hdim, xm_t);

    // 8) MLP up + GELU
    gemm_bf16<1><<<dim3(MLP_ / 128, LIMG_ / 128), 256, 0, stream>>>(
        xm_i, Pi(12), Pi(13), nullptr, nullptr, act_i, LIMG_, MLP_, Hdim);
    gemm_bf16<1><<<dim3(MLP_ / 128, LTXT_ / 128), 256, 0, stream>>>(
        xm_t, Pt(12), Pt(13), nullptr, nullptr, act_t, LTXT_, MLP_, Hdim);

    // 9) MLP down + gated residual -> d_out
    gemm_bf16<2><<<dim3(Hdim / 128, LIMG_ / 128), 256, 0, stream>>>(
        act_i, Pi(14), Pi(15), mod_i + 5 * Hdim, x1_i, out_img, LIMG_, Hdim, MLP_);
    gemm_bf16<2><<<dim3(Hdim / 128, LTXT_ / 128), 256, 0, stream>>>(
        act_t, Pt(14), Pt(15), mod_t + 5 * Hdim, x1_t, out_txt, LTXT_, Hdim, MLP_);
}
